// TreeLSTM_50319836839979
// MI455X (gfx1250) — matrix operand order repaired and checked by reading, not compiled, not run on hardware
//
#include <hip/hip_runtime.h>
#include <hip/hip_bf16.h>
#include <math.h>

typedef __attribute__((ext_vector_type(16))) _Float16 v16h;
typedef __attribute__((ext_vector_type(8)))  _Float16 v8h;
typedef __attribute__((ext_vector_type(8)))  float    v8f;

#define N_NODES 16384
#define IN_DIM  128
#define H_DIM   256
#define G4      1024   // 4*H
#define NVOCAB  100
#define MAXR    2752   // max padded non-leaf rows per level (level 6: 2731 -> 2752)

// ---------------------------------------------------------------- utilities
__global__ __launch_bounds__(256) void k_f32_to_f16(const float* __restrict__ s,
                                                    _Float16* __restrict__ d, int n) {
  int i = blockIdx.x * 256 + threadIdx.x;
  if (i < n) d[i] = (_Float16)s[i];
}

// normalize each embedding row (VOCAB x 128), emit f16
__global__ __launch_bounds__(128) void k_norm_embed(const float* __restrict__ emb,
                                                    _Float16* __restrict__ out) {
  __shared__ float red[128];
  int r = blockIdx.x, t = threadIdx.x;
  float v = emb[r * IN_DIM + t];
  red[t] = v * v;
  __syncthreads();
  for (int s = 64; s > 0; s >>= 1) {
    if (t < s) red[t] += red[t + s];
    __syncthreads();
  }
  float inv = rsqrtf(red[0]);
  out[r * IN_DIM + t] = (_Float16)(v * inv);
}

// x_f16[n][j] = normed_embed[node_ids[n]][j]
__global__ __launch_bounds__(256) void k_gather_x(const int* __restrict__ ids,
                                                  const _Float16* __restrict__ embn,
                                                  _Float16* __restrict__ x) {
  int i = blockIdx.x * 256 + threadIdx.x;          // over N*128
  int n = i >> 7, j = i & 127;
  x[i] = embn[ids[n] * IN_DIM + j];
}

// ---------------------------------------------------- fragment load helpers
// A frag (16x32 f16, MxK): lanes 0-15 row = lane, K {0..15};
// lanes 16-31 same rows, K {16..31}.  `p` pre-offset by (row*lda + hi*16).
__device__ __forceinline__ v16h load_a_frag(const _Float16* p) {
  v8h lo = *(const v8h*)(p);
  v8h hi = *(const v8h*)(p + 8);
  return __builtin_shufflevector(lo, hi, 0, 1, 2, 3, 4, 5, 6, 7,
                                 8, 9, 10, 11, 12, 13, 14, 15);
}
// B frag (32x16 f16, KxN): lane = column (lane&15); lanes>=16 take K 16..31.
// `p` pre-offset by (col*ldb + hi*16): one contiguous 32B chunk.
__device__ __forceinline__ v16h load_b_frag(const _Float16* p) {
  return *(const v16h*)(p);
}

// ------------------------------------------------- wave GEMM: C = A * B^T
// Per-wave register tile: 32 rows x 64 cols = 8 WMMA accumulators.
// Per K-step (32): 2 A frags + 4 B frags feed 8 WMMAs (0.75 frags/WMMA).
// grid = (M/32, 2), block = 256 (8 waves). Column slot = blockIdx.y*8 + wave,
// each slot covers 64 columns; 16 slots * 64 = 1024 = G4.
template <int K>
__global__ __launch_bounds__(256)
void k_gemm_tn(const _Float16* __restrict__ A, int lda,
               const _Float16* __restrict__ B, int ldb,
               const float* __restrict__ bias,
               float* __restrict__ C, int ldc) {
  const int wave = threadIdx.x >> 5;
  const int lane = threadIdx.x & 31;
  const int m0 = blockIdx.x * 32;
  const int n0 = (blockIdx.y * 8 + wave) * 64;
  const int rlo = lane & 15;
  const int hi  = lane >> 4;

  const _Float16* a0 = A + (size_t)(m0 + rlo) * lda + hi * 16;
  const _Float16* b0 = B + (size_t)(n0 + rlo) * ldb + hi * 16;

  v8f acc[8] = {};   // acc[r*4+c]: row tile r in {0,1}, col tile c in {0..3}
#pragma unroll
  for (int k0 = 0; k0 < K; k0 += 32) {
    v16h af[2], bf[4];
    af[0] = load_a_frag(a0 + k0);
    af[1] = load_a_frag(a0 + (size_t)16 * lda + k0);
#pragma unroll
    for (int c = 0; c < 4; ++c)
      bf[c] = load_b_frag(b0 + (size_t)(c * 16) * ldb + k0);
    if (k0 + 32 < K) {   // hint next A K-block into cache (global_prefetch_b8)
      __builtin_prefetch(a0 + k0 + 32, 0, 1);
    }
#pragma unroll
    for (int r = 0; r < 2; ++r)
#pragma unroll
      for (int c = 0; c < 4; ++c)
        acc[r * 4 + c] = __builtin_amdgcn_wmma_f32_16x16x32_f16(
            false, af[r], false, bf[c], (short)0, acc[r * 4 + c], false, false);
  }

  // C layout per tile: lane holds column rlo; rows v (+8 for upper lane half)
#pragma unroll
  for (int r = 0; r < 2; ++r) {
#pragma unroll
    for (int c = 0; c < 4; ++c) {
      float bv = bias ? bias[n0 + c * 16 + rlo] : 0.0f;
      float* crow = C + (size_t)(m0 + r * 16 + hi * 8) * ldc + n0 + c * 16 + rlo;
#pragma unroll
      for (int v = 0; v < 8; ++v) crow[(size_t)v * ldc] = acc[r * 4 + c][v] + bv;
    }
  }
}

// ------------------------------------ per-level child gather (contiguous kids)
// block = 256 (one column each), grid = padRows. Rows >= gemmRows write zeros.
__global__ __launch_bounds__(256)
void k_gather_children(const float* __restrict__ h, const float* __restrict__ c,
                       int nodeStart, int gemmRows,
                       _Float16* __restrict__ hs, float* __restrict__ cin) {
  int l = blockIdx.x, k = threadIdx.x;
  float hsum = 0.f, csum = 0.f, cnt = 0.f;
  if (l < gemmRows) {
    int n = nodeStart + l;
    int ch0 = 4 * n + 1;
#pragma unroll
    for (int j = 0; j < 4; ++j) {
      int ch = ch0 + j;
      if (ch < N_NODES) {
        hsum += h[(size_t)ch * H_DIM + k];
        csum += c[(size_t)ch * H_DIM + k];
        cnt += 1.0f;
      }
    }
  }
  hs[(size_t)l * H_DIM + k]  = (_Float16)hsum;
  cin[(size_t)l * H_DIM + k] = csum / fmaxf(cnt, 1.0f);
}

// ------------------------------------------------- per-level LSTM elementwise
__global__ __launch_bounds__(256)
void k_lstm_level(const float* __restrict__ gx, const float* __restrict__ gemm,
                  const float* __restrict__ cin, const float* __restrict__ b_hh,
                  float* __restrict__ h, float* __restrict__ c,
                  int nodeStart, int gemmRows) {
  int l = blockIdx.x, k = threadIdx.x;
  int n = nodeStart + l;
  size_t g = (size_t)n * G4 + k;
  float gi = gx[g]       + b_hh[k];
  float gf = gx[g + 256] + b_hh[k + 256];
  float gg = gx[g + 512] + b_hh[k + 512];
  float go = gx[g + 768] + b_hh[k + 768];
  float ci = 0.f;
  if (l < gemmRows) {
    size_t r = (size_t)l * G4 + k;
    gi += gemm[r];
    gf += gemm[r + 256];
    gg += gemm[r + 512];
    go += gemm[r + 768];
    ci = cin[(size_t)l * H_DIM + k];
  }
  float si = 1.0f / (1.0f + expf(-gi));
  float sf = 1.0f / (1.0f + expf(-gf));
  float so = 1.0f / (1.0f + expf(-go));
  float cc = sf * ci + si * tanhf(gg);
  float hh = so * tanhf(cc);
  h[(size_t)n * H_DIM + k] = hh;
  c[(size_t)n * H_DIM + k] = cc;
}

__global__ __launch_bounds__(512)
void k_write_out(const float* __restrict__ h, const float* __restrict__ c,
                 float* __restrict__ out) {
  int t = threadIdx.x;
  out[t] = (t < 256) ? h[t] : c[t - 256];
}

// --------------------------------------------------------------------- host
extern "C" void kernel_launch(void* const* d_in, const int* in_sizes, int n_in,
                              void* d_out, int out_size, void* d_ws, size_t ws_size,
                              hipStream_t stream) {
  (void)in_sizes; (void)n_in; (void)out_size; (void)ws_size;
  const int*   node_ids  = (const int*)d_in[0];
  const float* embedding = (const float*)d_in[4];
  const float* W_ih      = (const float*)d_in[5];
  const float* W_hh      = (const float*)d_in[6];
  const float* b_ih      = (const float*)d_in[7];
  const float* b_hh      = (const float*)d_in[8];
  float* out = (float*)d_out;

  char* w = (char*)d_ws;
  size_t off = 0;
  auto take = [&](size_t bytes) -> void* {
    void* p = w + off;
    off = (off + bytes + 255) & ~(size_t)255;
    return p;
  };
  _Float16* Wih_h = (_Float16*)take((size_t)G4 * IN_DIM * 2);
  _Float16* Whh_h = (_Float16*)take((size_t)G4 * H_DIM * 2);
  _Float16* embn  = (_Float16*)take((size_t)NVOCAB * IN_DIM * 2);
  _Float16* x_h   = (_Float16*)take((size_t)N_NODES * IN_DIM * 2);
  float*    gx    = (float*)take((size_t)N_NODES * G4 * 4);
  float*    hbuf  = (float*)take((size_t)N_NODES * H_DIM * 4);
  float*    cbuf  = (float*)take((size_t)N_NODES * H_DIM * 4);
  _Float16* hs    = (_Float16*)take((size_t)MAXR * H_DIM * 2);
  float*    cin   = (float*)take((size_t)MAXR * H_DIM * 4);
  float*    gemm  = (float*)take((size_t)MAXR * G4 * 4);

  // weight + embedding prep (parallel, level-independent)
  k_f32_to_f16<<<(G4 * IN_DIM + 255) / 256, 256, 0, stream>>>(W_ih, Wih_h, G4 * IN_DIM);
  k_f32_to_f16<<<(G4 * H_DIM + 255) / 256, 256, 0, stream>>>(W_hh, Whh_h, G4 * H_DIM);
  k_norm_embed<<<NVOCAB, 128, 0, stream>>>(embedding, embn);
  k_gather_x<<<(N_NODES * IN_DIM) / 256, 256, 0, stream>>>(node_ids, embn, x_h);

  // gx = x @ W_ih^T + b_ih   (16384x1024, K=128)
  k_gemm_tn<IN_DIM><<<dim3(N_NODES / 32, 2), 256, 0, stream>>>(
      x_h, IN_DIM, Wih_h, IN_DIM, b_ih, gx, G4);

  // bottom-up levels
  const int start[8] = {0, 1, 5, 21, 85, 341, 1365, 5461};
  const int pow4[8]  = {1, 4, 16, 64, 256, 1024, 4096, 16384};
  for (int d = 7; d >= 0; --d) {
    int s = start[d];
    int cnt = pow4[d];
    if (s + cnt > N_NODES) cnt = N_NODES - s;          // level 7 is partial: 10923
    int gemmRows = 4096 - s;                           // nodes <= 4095 have children
    if (gemmRows < 0) gemmRows = 0;
    if (gemmRows > cnt) gemmRows = cnt;
    int padRows = (gemmRows + 31) & ~31;               // pad to 32-row wave tile

    if (gemmRows > 0) {
      k_gather_children<<<padRows, 256, 0, stream>>>(hbuf, cbuf, s, gemmRows, hs, cin);
      // gemm = hs @ W_hh^T  (padRows x 1024, K=256)
      k_gemm_tn<H_DIM><<<dim3(padRows / 32, 2), 256, 0, stream>>>(
          hs, H_DIM, Whh_h, H_DIM, nullptr, gemm, G4);
    }
    k_lstm_level<<<cnt, 256, 0, stream>>>(gx, gemm, cin, b_hh, hbuf, cbuf, s, gemmRows);
  }

  k_write_out<<<1, 512, 0, stream>>>(hbuf, cbuf, out);
}
